// MultiHeadAttentionWithWavelet_81819126989058
// MI455X (gfx1250) — compile-verified, matrix operand order checked by reading, not run on hardware
//
#include <hip/hip_runtime.h>
#include <hip/hip_bf16.h>

typedef __attribute__((ext_vector_type(16))) __bf16 v16bf;
typedef __attribute__((ext_vector_type(8)))  float  v8f;

union FragBF {
  v16bf v;
  uint4 q[2];
  unsigned int u[8];
  unsigned short h[16];
};

__device__ __forceinline__ unsigned short f2bf(float f) {
  unsigned int u = __float_as_uint(f);
  u = (u + 0x7FFFu + ((u >> 16) & 1u)) >> 16;   // round-to-nearest-even
  return (unsigned short)u;
}

// ---------------------------------------------------------------- constants
// db4 filters, pre-reversed (f[i] = DEC_*[7-i]) for the valid-conv inner loop
__constant__ float FLO_R[8] = {
   0.23037781330885523f,  0.7148465705525415f,   0.6308807679295904f,
  -0.02798376941698385f, -0.18703481171888114f,  0.030841381835986965f,
   0.032883011666982945f,-0.010597401784997278f };
__constant__ float FHI_R[8] = {
  -0.010597401784997278f,-0.032883011666982945f, 0.030841381835986965f,
   0.18703481171888114f, -0.02798376941698385f, -0.6308807679295904f,
   0.7148465705525415f,  -0.23037781330885523f };

#define Bsz 4
#define Ssz 2048
#define Hd  1024
#define NH  8
#define DK  128
#define WA  67
#define WP  80

// ---------------------------------------------------------------- kernel 1
__global__ void cvt_f32_bf16(const float* __restrict__ src,
                             unsigned short* __restrict__ dst, int n) {
  int i = blockIdx.x * blockDim.x + threadIdx.x;
  if (i < n) dst[i] = f2bf(src[i]);
}

// ---------------------------------------------------------------- kernel 2
// y[m,n] = sum_k xb[m,k]*wb[n,k];  m in [0,8192), n,k in [0,1024)
// MODE 0: store bf16 *scale (q), 1: store bf16 (k), 2: store f32 (v)
// output layout (b,h,s,d): ((b*NH+h)*Ssz+s)*DK + d
template <int MODE>
__global__ void __launch_bounds__(256) gemm_qkv(
    const unsigned short* __restrict__ xb,
    const unsigned short* __restrict__ wb,
    void* __restrict__ out, float scale) {
  const int wave = threadIdx.x >> 5;
  const int lane = threadIdx.x & 31;
  const int gw   = blockIdx.x * 8 + wave;
  const int nt   = gw & 63;        // 64 n-tiles
  const int mt   = gw >> 6;        // 512 m-tiles
  const int lo   = lane & 15;
  const int hi8  = (lane >> 4) * 8;

  const unsigned short* arow = xb + (mt * 16 + lo) * Hd;   // A: row = M
  const unsigned short* brow = wb + (nt * 16 + lo) * Hd;   // B: row = N (W[n,k])

  v8f acc = {0.f, 0.f, 0.f, 0.f, 0.f, 0.f, 0.f, 0.f};
  for (int kk = 0; kk < Hd; kk += 32) {
    FragBF a, b;
    a.q[0] = *(const uint4*)(arow + kk + hi8);
    a.q[1] = *(const uint4*)(arow + kk + 16 + hi8);
    b.q[0] = *(const uint4*)(brow + kk + hi8);
    b.q[1] = *(const uint4*)(brow + kk + 16 + hi8);
    acc = __builtin_amdgcn_wmma_f32_16x16x32_bf16(
        false, a.v, false, b.v, (short)0, acc, false, false);
  }

  const int g = lane >> 4;
#pragma unroll
  for (int r = 0; r < 8; ++r) {
    int m = mt * 16 + r + 8 * g;
    int n = nt * 16 + lo;
    int b_ = m >> 11, s = m & (Ssz - 1);
    int h  = n >> 7,  d = n & (DK - 1);
    int idx = ((b_ * NH + h) * Ssz + s) * DK + d;
    float val = acc[r];
    if (MODE == 0)      ((unsigned short*)out)[idx] = f2bf(val * scale);
    else if (MODE == 1) ((unsigned short*)out)[idx] = f2bf(val);
    else                ((float*)out)[idx] = val;
  }
}

// ---------------------------------------------------------------- kernel 3
// db4 DWT along dk: cA (f32) -> out2 (raw (B,H,S,67) order),
// cD -> bf16, stored transposed (bh, w[0..80), s) and zero padded w>=67.
__global__ void dwt_db4_k(const float* __restrict__ vf,
                          float* __restrict__ out2,
                          unsigned short* __restrict__ cdt) {
  __shared__ float sv[DK];
  const int row = blockIdx.x;          // bh*Ssz + s, 65536 rows
  const int t   = threadIdx.x;         // 0..127
  sv[t] = vf[row * DK + t];
  __syncthreads();
  if (t < WP) {
    const int bh = row >> 11, s = row & (Ssz - 1);
    unsigned short dq = 0;
    if (t < WA) {
      float a = 0.f, d = 0.f;
#pragma unroll
      for (int i = 0; i < 8; ++i) {
        int p = 2 * t + i + 1;                                   // padded index
        int src = (p < 7) ? (6 - p) : (p < DK + 7 ? p - 7 : 2 * DK + 6 - p);
        float x = sv[src];
        a += x * FLO_R[i];
        d += x * FHI_R[i];
      }
      out2[row * WA + t] = a;
      dq = f2bf(d);
    }
    cdt[(bh * WP + t) * Ssz + s] = dq;
  }
}

// ---------------------------------------------------------------- kernel 4
// Flash attention with wavelet values: out1 = softmax(QK^T) @ cD
// 32 keys per iteration: P tile is a full 16x32 bf16 A-fragment (no padding),
// softmax lane-reductions amortized over 32 keys.
__global__ void __launch_bounds__(256, 1) flash_wav(
    const unsigned short* __restrict__ qb,
    const unsigned short* __restrict__ kb,
    const unsigned short* __restrict__ cdt,
    float* __restrict__ out1) {
  __shared__ __align__(16) unsigned short pbuf[8][16 * 32];
  const int wave = threadIdx.x >> 5;
  const int lane = threadIdx.x & 31;
  const int gw   = blockIdx.x * 8 + wave;
  const int bh   = gw >> 7;          // 32 (b,h) pairs
  const int qt   = gw & 127;         // 128 q-tiles of 16 rows
  const int lo   = lane & 15;
  const int g    = lane >> 4;
  const int hi8  = g * 8;
  unsigned short* pw = &pbuf[wave][0];

  // Q A-fragments: 16 rows x 128 features (4 fragments of K=32)
  FragBF qA[4];
  const unsigned short* qrow = qb + (bh * Ssz + qt * 16 + lo) * DK;
#pragma unroll
  for (int f = 0; f < 4; ++f) {
    qA[f].q[0] = *(const uint4*)(qrow + f * 32 + hi8);
    qA[f].q[1] = *(const uint4*)(qrow + f * 32 + 16 + hi8);
  }

  float mi[8], li[8];
#pragma unroll
  for (int r = 0; r < 8; ++r) { mi[r] = -3.0e38f; li[r] = 0.f; }
  v8f acc[5];
#pragma unroll
  for (int t = 0; t < 5; ++t)
#pragma unroll
    for (int r = 0; r < 8; ++r) acc[t][r] = 0.f;

  const unsigned short* kbase = kb + bh * Ssz * DK;
  const unsigned short* cbase = cdt + bh * WP * Ssz;

  for (int kt = 0; kt < 64; ++kt) {            // 32 keys per iteration
    // ---- S = Q K^T : two 16x16 f32 tiles, keys kt*32 .. +31
    const unsigned short* krow0 = kbase + (kt * 32 + lo) * DK;
    const unsigned short* krow1 = krow0 + 16 * DK;
    v8f s0 = {0.f, 0.f, 0.f, 0.f, 0.f, 0.f, 0.f, 0.f};
    v8f s1 = {0.f, 0.f, 0.f, 0.f, 0.f, 0.f, 0.f, 0.f};
#pragma unroll
    for (int f = 0; f < 4; ++f) {
      FragBF k0, k1;
      k0.q[0] = *(const uint4*)(krow0 + f * 32 + hi8);
      k0.q[1] = *(const uint4*)(krow0 + f * 32 + 16 + hi8);
      k1.q[0] = *(const uint4*)(krow1 + f * 32 + hi8);
      k1.q[1] = *(const uint4*)(krow1 + f * 32 + 16 + hi8);
      s0 = __builtin_amdgcn_wmma_f32_16x16x32_bf16(
          false, qA[f].v, false, k0.v, (short)0, s0, false, false);
      s1 = __builtin_amdgcn_wmma_f32_16x16x32_bf16(
          false, qA[f].v, false, k1.v, (short)0, s1, false, false);
    }
    // prefetch next key tile while softmax VALU/trans work runs
    if (kt < 63) {
      __builtin_prefetch(krow0 + 32 * DK, 0, 3);
      __builtin_prefetch(krow0 + 48 * DK, 0, 3);
    }

    // ---- online softmax; row m = r + 8g lives on lanes [g*16, g*16+15]
    float corr[8];
#pragma unroll
    for (int r = 0; r < 8; ++r) {
      float x = fmaxf(s0[r], s1[r]);           // combine both tiles first
#pragma unroll
      for (int off = 8; off >= 1; off >>= 1)
        x = fmaxf(x, __shfl_xor(x, off, 16));
      float nm = fmaxf(mi[r], x);
      corr[r]  = __expf(mi[r] - nm);
      mi[r]    = nm;
      float p0 = __expf(s0[r] - nm);
      float p1 = __expf(s1[r] - nm);
      // P (bf16) into per-wave LDS tile: cols lo (keys 0..15), lo+16 (16..31)
      pw[(r + 8 * g) * 32 + lo]      = f2bf(p0);
      pw[(r + 8 * g) * 32 + 16 + lo] = f2bf(p1);
      float rs = p0 + p1;
#pragma unroll
      for (int off = 8; off >= 1; off >>= 1)
        rs += __shfl_xor(rs, off, 16);
      li[r] = li[r] * corr[r] + rs;
    }
#pragma unroll
    for (int t = 0; t < 5; ++t)
#pragma unroll
      for (int r = 0; r < 8; ++r) acc[t][r] *= corr[r];

    // ---- P: C-layout -> bf16 A-layout (16x32, fully populated)
    asm volatile("s_wait_dscnt 0x0" ::: "memory");
    FragBF pA;
    pA.q[0] = *(const uint4*)(pw + lo * 32 + hi8);
    pA.q[1] = *(const uint4*)(pw + lo * 32 + 16 + hi8);
    asm volatile("s_wait_dscnt 0x0" ::: "memory");

    // ---- acc += P @ cD tile (K=32 keys, 5 n-tiles of 16 over padded 80)
#pragma unroll
    for (int t = 0; t < 5; ++t) {
      FragBF cf;
      cf.q[0] = *(const uint4*)(cbase + (t * 16 + lo) * Ssz + kt * 32 + hi8);
      cf.q[1] = *(const uint4*)(cbase + (t * 16 + lo) * Ssz + kt * 32 + 16 + hi8);
      acc[t] = __builtin_amdgcn_wmma_f32_16x16x32_bf16(
          false, pA.v, false, cf.v, (short)0, acc[t], false, false);
    }
  }

  // ---- normalize and store out1[b][s][h*67 + w]
  const int b_ = bh >> 3, h = bh & 7;
#pragma unroll
  for (int r = 0; r < 8; ++r) {
    float inv = 1.0f / li[r];
    int s = qt * 16 + r + 8 * g;
    float* orow = out1 + (b_ * Ssz + s) * (NH * WA) + h * WA;
#pragma unroll
    for (int t = 0; t < 5; ++t) {
      int w = t * 16 + lo;
      if (w < WA) orow[w] = acc[t][r] * inv;
    }
  }
}

// ---------------------------------------------------------------- launch
extern "C" void kernel_launch(void* const* d_in, const int* in_sizes, int n_in,
                              void* d_out, int out_size, void* d_ws, size_t ws_size,
                              hipStream_t stream) {
  const float* x  = (const float*)d_in[0];
  const float* Wq = (const float*)d_in[1];
  const float* Wk = (const float*)d_in[2];
  const float* Wv = (const float*)d_in[3];

  char* ws = (char*)d_ws;
  unsigned short* xb  = (unsigned short*)(ws + 0);          // 16.78 MB
  unsigned short* wqb = (unsigned short*)(ws + 16777216);   //  2.10 MB
  unsigned short* wkb = (unsigned short*)(ws + 18874368);   //  2.10 MB
  unsigned short* wvb = (unsigned short*)(ws + 20971520);   //  2.10 MB
  unsigned short* qb  = (unsigned short*)(ws + 23068672);   // 16.78 MB
  unsigned short* kbf = (unsigned short*)(ws + 39845888);   // 16.78 MB
  float*          vf  = (float*)         (ws + 56623104);   // 33.55 MB
  unsigned short* cdt = (unsigned short*)(ws + 90177536);   // 10.49 MB

  float* out1 = (float*)d_out;
  float* out2 = out1 + Bsz * Ssz * NH * WA;   // 4,390,912 floats each

  const int nx = Bsz * Ssz * Hd;   // 8388608
  const int nw = Hd * Hd;          // 1048576
  cvt_f32_bf16<<<(nx + 255) / 256, 256, 0, stream>>>(x,  xb,  nx);
  cvt_f32_bf16<<<(nw + 255) / 256, 256, 0, stream>>>(Wq, wqb, nw);
  cvt_f32_bf16<<<(nw + 255) / 256, 256, 0, stream>>>(Wk, wkb, nw);
  cvt_f32_bf16<<<(nw + 255) / 256, 256, 0, stream>>>(Wv, wvb, nw);

  const float scale = 0.08838834764831845f;   // 1/sqrt(128)
  gemm_qkv<0><<<4096, 256, 0, stream>>>(xb, wqb, qb,  scale);
  gemm_qkv<1><<<4096, 256, 0, stream>>>(xb, wkb, kbf, 1.0f);
  gemm_qkv<2><<<4096, 256, 0, stream>>>(xb, wvb, vf,  1.0f);

  dwt_db4_k<<<Bsz * NH * Ssz, 128, 0, stream>>>(vf, out2, cdt);

  flash_wav<<<512, 256, 0, stream>>>(qb, kbf, cdt, out1);
}
